// FAN_87771951661628
// MI455X (gfx1250) — compile-verified, matrix operand order checked by reading, not run on hardware
//
#include <hip/hip_runtime.h>
#include <hip/hip_bf16.h>

// ---------------------------------------------------------------------------
// Types for CDNA5 WMMA (wave32): D(16x16 f32) = A(16x32 bf16) x B(32x16 bf16) + C
// ---------------------------------------------------------------------------
typedef __attribute__((ext_vector_type(16))) __bf16 v16bf;
typedef __attribute__((ext_vector_type(8)))  float  v8f;

enum { GF_RELU = 1, GF_ADD = 2 };

static __device__ __forceinline__ float sigmoidf_dev(float v) {
    return 1.0f / (1.0f + __expf(-v));
}

// ---------------------------------------------------------------------------
// f32 -> bf16 elementwise convert
// ---------------------------------------------------------------------------
__global__ __launch_bounds__(256) void f2bf_kernel(const float* __restrict__ s,
                                                   __bf16* __restrict__ d, int n) {
    int i = blockIdx.x * 256 + threadIdx.x;
    if (i < n) d[i] = (__bf16)s[i];
}

// ---------------------------------------------------------------------------
// Global average pool per channel: g[c] = mean over N spatial elems
// ---------------------------------------------------------------------------
__global__ __launch_bounds__(256) void gap_kernel(const float* __restrict__ X,
                                                  float* __restrict__ g, int N) {
    __shared__ float red[256];
    const int c = blockIdx.x;
    float s = 0.0f;
    for (int i = threadIdx.x; i < N; i += 256) s += X[(size_t)c * N + i];
    red[threadIdx.x] = s;
    __syncthreads();
    for (int st = 128; st > 0; st >>= 1) {
        if (threadIdx.x < st) red[threadIdx.x] += red[threadIdx.x + st];
        __syncthreads();
    }
    if (threadIdx.x == 0) g[c] = red[0] / (float)N;
}

// ---------------------------------------------------------------------------
// atten[o] = sigmoid( dot(Wt[o,:], g) )   (1x1 conv on 1x1 spatial, no bias)
// ---------------------------------------------------------------------------
__global__ __launch_bounds__(256) void gemv_sigmoid_kernel(const float* __restrict__ Wt,
                                                           const float* __restrict__ g,
                                                           float* __restrict__ out, int n) {
    __shared__ float red[256];
    const int o = blockIdx.x;
    float s = 0.0f;
    for (int i = threadIdx.x; i < n; i += 256) s += Wt[(size_t)o * n + i] * g[i];
    red[threadIdx.x] = s;
    __syncthreads();
    for (int st = 128; st > 0; st >>= 1) {
        if (threadIdx.x < st) red[threadIdx.x] += red[threadIdx.x + st];
        __syncthreads();
    }
    if (threadIdx.x == 0) out[o] = sigmoidf_dev(red[0]);
}

// ---------------------------------------------------------------------------
// armIn[c,n] = bf16( x[c,n] * (1 + atten[c]) )     (fsm residual-gate input)
// ---------------------------------------------------------------------------
__global__ __launch_bounds__(256) void scale_arm_kernel(const float* __restrict__ X,
                                                        const float* __restrict__ att,
                                                        __bf16* __restrict__ out,
                                                        int N, int total) {
    int idx = blockIdx.x * 256 + threadIdx.x;
    if (idx >= total) return;
    int c = idx / N;
    out[idx] = (__bf16)(X[idx] * (1.0f + att[c]));
}

// ---------------------------------------------------------------------------
// 2x bilinear upsample (align_corners=False, half-pixel, edge clamp).
// Also writes bf16(2*v) into U2 (the upper half of the off-conv concat input).
// ---------------------------------------------------------------------------
__global__ __launch_bounds__(256) void upsample2x_kernel(const float* __restrict__ S,
                                                         float* __restrict__ U,
                                                         __bf16* __restrict__ U2,
                                                         int C, int h, int w) {
    const int H = 2 * h, W = 2 * w, N = H * W;
    int idx = blockIdx.x * 256 + threadIdx.x;
    if (idx >= C * N) return;
    const int c = idx / N, n = idx - c * N;
    const int y = n / W, x = n - y * W;
    const float sy = y * 0.5f - 0.25f;
    const float sx = x * 0.5f - 0.25f;
    const float fy = floorf(sy), fx = floorf(sx);
    const float wy1 = sy - fy, wx1 = sx - fx;
    int y0 = (int)fy, x0 = (int)fx;
    int y1 = y0 + 1, x1 = x0 + 1;
    y0 = min(max(y0, 0), h - 1); y1 = min(max(y1, 0), h - 1);
    x0 = min(max(x0, 0), w - 1); x1 = min(max(x1, 0), w - 1);
    const float* ch = S + (size_t)c * h * w;
    const float v = (1.0f - wy1) * ((1.0f - wx1) * ch[y0 * w + x0] + wx1 * ch[y0 * w + x1])
                  +          wy1 * ((1.0f - wx1) * ch[y1 * w + x0] + wx1 * ch[y1 * w + x1]);
    U[idx] = v;
    if (U2) U2[idx] = (__bf16)(2.0f * v);
}

// ---------------------------------------------------------------------------
// Modulated deformable sampling: builds B[(g*32+cg)*9+k][n] (bf16) =
//   bilinear(feat_up[g*32+cg], y+ky-1+off_y, x+kx-1+off_x) * sigmoid(mask)
// om layout: [0:144) offsets ({g,d,k}: g*18+d*9+k), [144:216) mask (g*9+k).
// ---------------------------------------------------------------------------
__global__ __launch_bounds__(256) void mdcn_sample_kernel(const float* __restrict__ U,
                                                          const float* __restrict__ om,
                                                          __bf16* __restrict__ B,
                                                          int H, int W) {
    const int N = H * W;
    const int n = blockIdx.x * 256 + threadIdx.x;
    if (n >= N) return;
    const int gk = blockIdx.y;            // 0..71 = g*9+k
    const int g = gk / 9, k = gk - g * 9;
    const int y = n / W, x = n - y * W;
    const float offy = om[(size_t)(g * 18 + k) * N + n];
    const float offx = om[(size_t)(g * 18 + 9 + k) * N + n];
    const float m = sigmoidf_dev(om[(size_t)(144 + gk) * N + n]);
    const float py = (float)(y + k / 3 - 1) + offy;
    const float px = (float)(x + (k % 3) - 1) + offx;
    const float fy = floorf(py), fx = floorf(px);
    const int y0 = (int)fy, x0 = (int)fx;
    const float wy1 = py - fy, wy0 = 1.0f - wy1;
    const float wx1 = px - fx, wx0 = 1.0f - wx1;
    const bool vy0 = (y0 >= 0) && (y0 < H);
    const bool vy1 = (y0 + 1 >= 0) && (y0 + 1 < H);
    const bool vx0 = (x0 >= 0) && (x0 < W);
    const bool vx1 = (x0 + 1 >= 0) && (x0 + 1 < W);
    const float w00 = (vy0 && vx0) ? wy0 * wx0 * m : 0.0f;
    const float w01 = (vy0 && vx1) ? wy0 * wx1 * m : 0.0f;
    const float w10 = (vy1 && vx0) ? wy1 * wx0 * m : 0.0f;
    const float w11 = (vy1 && vx1) ? wy1 * wx1 * m : 0.0f;
    const int yc0 = min(max(y0, 0), H - 1);
    const int yc1 = min(max(y0 + 1, 0), H - 1);
    const int xc0 = min(max(x0, 0), W - 1);
    const int xc1 = min(max(x0 + 1, 0), W - 1);
    const int i00 = yc0 * W + xc0, i01 = yc0 * W + xc1;
    const int i10 = yc1 * W + xc0, i11 = yc1 * W + xc1;
#pragma unroll 4
    for (int cg = 0; cg < 32; ++cg) {
        const int c = g * 32 + cg;
        const float* ch = U + (size_t)c * N;
        const float v = w00 * ch[i00] + w01 * ch[i01] + w10 * ch[i10] + w11 * ch[i11];
        B[((size_t)c * 9 + k) * N + n] = (__bf16)v;
    }
}

// ---------------------------------------------------------------------------
// WMMA GEMM:  out[M,N] = A[M,K](bf16) x B[K,N] + bias, optional relu/residual.
// ksize==1 -> X is [K][N]; ksize==3 -> implicit im2col of X=[K/9][H][W], pad=1
// (k-order ci*9 + ky*3 + kx matches row-major OIHW weights).
//
// Block = 256 threads = 8 waves; block tile 128x128. Each wave owns a 32x64
// patch: 2 A-fragments x 4 B-fragments -> 8 v_wmma per K-step (fragment reuse
// 2-4x). K stepped by 32 through LDS.
//
// A tile staged with CDNA5 GLOBAL_LOAD_ASYNC_TO_LDS_B128 (inline asm; LDS
// offset = low 32 bits of the generic shared-pointer, per ISA flat-LDS
// aperture mapping), synchronized with s_wait_asynccnt before the barrier.
// Ragged M rows fall back to ds-store of zeros.
//
// bf16 A-fragment layout (ISA 7.12.2): lane<16 -> M=lane, K in
// {kb..kb+7, kb+16..kb+23}, kb=8*(lane>=16); B assumed mirrored (lane&15 ->
// column). C/D: VGPR v: lane<16 -> M=v, N=lane; lane>=16 -> M=8+v, N=lane-16.
// ---------------------------------------------------------------------------
__global__ __launch_bounds__(256) void wmma_gemm_kernel(
    const __bf16* __restrict__ A, const __bf16* __restrict__ X,
    const float* __restrict__ bias, const float* __restrict__ addend,
    float* __restrict__ outF, __bf16* __restrict__ outB,
    int M, int K, int N, int H, int W, int ksize, int flags)
{
    __shared__ __align__(16) __bf16 As[128][32];   // [m][k]
    __shared__ __align__(16) __bf16 Bs[128][32];   // [n][k]  (transposed)

    const int t    = threadIdx.x;
    const int n0   = blockIdx.x * 128;
    const int m0   = blockIdx.y * 128;
    const int lane = t & 31;
    const int wv   = t >> 5;
    const int mq   = wv & 3;          // wave's 32-row band
    const int nq   = wv >> 2;         // wave's 64-col band
    const int r    = lane & 15;
    const int half = lane >> 4;
    const int kb   = half * 8;

    v8f acc[2][4] = {};

    // A staging: thread -> (row = t>>1, kc = (t&1)*16): 16 elems = 2 x 16B
    const int arow = t >> 1;          // 0..127
    const int akc  = (t & 1) * 16;    // 0 or 16
    // B staging: thread -> (n = t&127, kc = (t>>7)*16): 16 gathered elems
    const int bnl  = t & 127;         // 0..127
    const int bkc  = (t >> 7) * 16;   // 0 or 16
    const int gnB  = n0 + bnl;
    const int by   = gnB / W;
    const int bx   = gnB - by * W;

    for (int k0 = 0; k0 < K; k0 += 32) {
        // ---- stage A tile: async global->LDS (ASYNCcnt-tracked DMA path) ----
        {
            const int gm = m0 + arow;
            __bf16* lp = &As[arow][akc];
            if (gm < M) {
                const __bf16* gp = A + (size_t)gm * K + k0 + akc;
                __builtin_prefetch(gp + 32, 0, 1);      // next K-step, L2 pull
                const unsigned lds_off = (unsigned)(unsigned long long)(uintptr_t)lp;
                const unsigned long long ga = (unsigned long long)(uintptr_t)gp;
                asm volatile(
                    "global_load_async_to_lds_b128 %0, %1, off\n\t"
                    "global_load_async_to_lds_b128 %0, %1, off offset:16"
                    :: "v"(lds_off), "v"(ga) : "memory");
            } else {
                uint4 z = make_uint4(0u, 0u, 0u, 0u);
                *reinterpret_cast<uint4*>(lp)     = z;
                *reinterpret_cast<uint4*>(lp + 8) = z;
            }
        }
        // ---- stage B tile (implicit im2col for 3x3) ----
        if (ksize == 1) {
#pragma unroll
            for (int i = 0; i < 16; ++i) {
                const int gk = k0 + bkc + i;
                Bs[bnl][bkc + i] = X[(size_t)gk * N + gnB];
            }
        } else {
#pragma unroll
            for (int i = 0; i < 16; ++i) {
                const int gk = k0 + bkc + i;
                const int ci = gk / 9;
                const int rr = gk - ci * 9;
                const int sy = by + rr / 3 - 1;
                const int sx = bx + (rr % 3) - 1;
                __bf16 v = (__bf16)0.0f;
                if (sy >= 0 && sy < H && sx >= 0 && sx < W)
                    v = X[((size_t)ci * H + sy) * W + sx];
                Bs[bnl][bkc + i] = v;
            }
        }
        asm volatile("s_wait_asynccnt 0x0" ::: "memory");
        __syncthreads();

        // ---- fragments from LDS (each 16 bf16 -> two ds_load_b128) ----
        v16bf a[2], b[4];
#pragma unroll
        for (int mi = 0; mi < 2; ++mi) {
            const int ar = mq * 32 + mi * 16 + r;
#pragma unroll
            for (int i = 0; i < 8; ++i) {
                a[mi][i]     = As[ar][kb + i];
                a[mi][8 + i] = As[ar][kb + 16 + i];
            }
        }
#pragma unroll
        for (int ni = 0; ni < 4; ++ni) {
            const int bc = nq * 64 + ni * 16 + r;
#pragma unroll
            for (int i = 0; i < 8; ++i) {
                b[ni][i]     = Bs[bc][kb + i];
                b[ni][8 + i] = Bs[bc][kb + 16 + i];
            }
        }
        // ---- 8 WMMAs, full fragment reuse ----
#pragma unroll
        for (int mi = 0; mi < 2; ++mi)
#pragma unroll
            for (int ni = 0; ni < 4; ++ni)
                acc[mi][ni] = __builtin_amdgcn_wmma_f32_16x16x32_bf16(
                    false, a[mi], false, b[ni], (short)0, acc[mi][ni], false, false);
        __syncthreads();
    }

    // ---- epilogue: bias / relu / residual add, f32 and/or bf16 stores ----
#pragma unroll
    for (int mi = 0; mi < 2; ++mi) {
#pragma unroll
        for (int v = 0; v < 8; ++v) {
            const int gm = m0 + mq * 32 + mi * 16 + half * 8 + v;
            if (gm >= M) continue;
            const float bv = bias ? bias[gm] : 0.0f;
#pragma unroll
            for (int ni = 0; ni < 4; ++ni) {
                const int gn = n0 + nq * 64 + ni * 16 + r;
                float val = acc[mi][ni][v] + bv;
                if (flags & GF_RELU) val = fmaxf(val, 0.0f);
                if (flags & GF_ADD)  val += addend[(size_t)gm * N + gn];
                if (outF) outF[(size_t)gm * N + gn] = val;
                if (outB) outB[(size_t)gm * N + gn] = (__bf16)val;
            }
        }
    }
}

// ===========================================================================
// Host orchestration
// ===========================================================================
extern "C" void kernel_launch(void* const* d_in, const int* in_sizes, int n_in,
                              void* d_out, int out_size, void* d_ws, size_t ws_size,
                              hipStream_t stream)
{
    (void)in_sizes; (void)n_in; (void)out_size; (void)ws_size;

    const float* c2 = (const float*)d_in[0];
    const float* c3 = (const float*)d_in[1];
    const float* c4 = (const float*)d_in[2];
    const float* c5 = (const float*)d_in[3];
    const float* lat_w = (const float*)d_in[31];
    const float* lat_b = (const float*)d_in[32];

    struct Lv {
        const float *att_w, *fsm_w, *off_w, *om_w, *om_b, *dcn_w, *dcn_b, *out_w, *out_b;
        const float* feat_l; int Cin, H, W;
    } lv[3];
    for (int i = 0; i < 3; ++i) {
        const int b = 4 + i * 9;
        lv[i].att_w = (const float*)d_in[b + 0];
        lv[i].fsm_w = (const float*)d_in[b + 1];
        lv[i].off_w = (const float*)d_in[b + 2];
        lv[i].om_w  = (const float*)d_in[b + 3];
        lv[i].om_b  = (const float*)d_in[b + 4];
        lv[i].dcn_w = (const float*)d_in[b + 5];
        lv[i].dcn_b = (const float*)d_in[b + 6];
        lv[i].out_w = (const float*)d_in[b + 7];
        lv[i].out_b = (const float*)d_in[b + 8];
    }
    lv[0].feat_l = c4; lv[0].Cin = 1024; lv[0].H = 32;  lv[0].W = 32;
    lv[1].feat_l = c3; lv[1].Cin = 512;  lv[1].H = 64;  lv[1].W = 64;
    lv[2].feat_l = c2; lv[2].Cin = 256;  lv[2].H = 128; lv[2].W = 128;

    // output regions: (p2, p3, p4, p5) flat in return order
    float* p2 = (float*)d_out;
    float* p3 = p2 + (size_t)256 * 128 * 128;
    float* p4 = p3 + (size_t)256 * 64 * 64;
    float* p5 = p4 + (size_t)256 * 32 * 32;
    float* pouts[3] = { p4, p3, p2 };

    // workspace bump allocator (256B aligned)
    size_t off = 0;
    auto alloc = [&](size_t bytes) -> void* {
        off = (off + 255) & ~(size_t)255;
        void* p = (char*)d_ws + off;
        off += bytes;
        return p;
    };

    __bf16* c5bf  = (__bf16*)alloc((size_t)2048 * 256 * 2);
    __bf16* latwB = (__bf16*)alloc((size_t)256 * 2048 * 2);
    __bf16 *fsmB[3], *offB[3], *omWB[3], *dcnWB[3], *outWB[3];
    for (int i = 0; i < 3; ++i) {
        fsmB[i]  = (__bf16*)alloc((size_t)256 * lv[i].Cin * 2);
        offB[i]  = (__bf16*)alloc((size_t)256 * 512 * 2);
        omWB[i]  = (__bf16*)alloc((size_t)216 * 2304 * 2);
        dcnWB[i] = (__bf16*)alloc((size_t)256 * 2304 * 2);
        outWB[i] = (__bf16*)alloc((size_t)256 * 2304 * 2);
    }
    float*  gapv    = (float*)alloc(2048 * 4);
    float*  attv    = (float*)alloc(2048 * 4);
    __bf16* armIn   = (__bf16*)alloc((size_t)1024 * 1024 * 4 * 2); // 4.2M elems max
    float*  featUp  = (float*)alloc((size_t)256 * 16384 * 4);
    __bf16* offIn   = (__bf16*)alloc((size_t)512 * 16384 * 2);
    __bf16* offFeat = (__bf16*)alloc((size_t)256 * 16384 * 2);
    float*  omOut   = (float*)alloc((size_t)216 * 16384 * 4);
    __bf16* sampB   = (__bf16*)alloc((size_t)2304 * 16384 * 2);
    float*  featArm = (float*)alloc((size_t)256 * 16384 * 4);
    float*  fa32[3];
    fa32[0] = (float*)alloc((size_t)256 * 1024 * 4);
    fa32[1] = (float*)alloc((size_t)256 * 4096 * 4);
    fa32[2] = (float*)alloc((size_t)256 * 16384 * 4);
    __bf16* fabf = (__bf16*)alloc((size_t)256 * 16384 * 2);

    auto conv2bf = [&](const float* s, __bf16* d, size_t n) {
        f2bf_kernel<<<(unsigned)((n + 255) / 256), 256, 0, stream>>>(s, d, (int)n);
    };
    conv2bf(c5, c5bf, (size_t)2048 * 256);
    conv2bf(lat_w, latwB, (size_t)256 * 2048);
    for (int i = 0; i < 3; ++i) {
        conv2bf(lv[i].fsm_w, fsmB[i],  (size_t)256 * lv[i].Cin);
        conv2bf(lv[i].off_w, offB[i],  (size_t)256 * 512);
        conv2bf(lv[i].om_w,  omWB[i],  (size_t)216 * 2304);
        conv2bf(lv[i].dcn_w, dcnWB[i], (size_t)256 * 2304);
        conv2bf(lv[i].out_w, outWB[i], (size_t)256 * 2304);
    }

    auto gemm = [&](const __bf16* A, const __bf16* X, const float* bias,
                    const float* addend, float* oF, __bf16* oB,
                    int M, int K, int H, int W, int ks, int flags) {
        const int N = H * W;
        dim3 g(N / 128, (M + 127) / 128);
        wmma_gemm_kernel<<<g, 256, 0, stream>>>(A, X, bias, addend, oF, oB,
                                                M, K, N, H, W, ks, flags);
    };

    // ---- p5 = lat_w @ c5 + lat_b  (1x1, 2048 -> 256, 16x16) ----
    gemm(latwB, c5bf, lat_b, nullptr, p5, nullptr, 256, 2048, 16, 16, 1, 0);

    // ---- three FeatureAlign_V2 levels (coarse -> fine) ----
    for (int i = 0; i < 3; ++i) {
        const int H = lv[i].H, W = lv[i].W, N = H * W, Cin = lv[i].Cin;
        const float* fs = (i == 0) ? p5 : fa32[i - 1];   // feat_s (f32)

        // bilinear upsample -> featUp (f32); 2*feat_up bf16 -> offIn[256:512]
        {
            const int tot = 256 * N;
            upsample2x_kernel<<<(tot + 255) / 256, 256, 0, stream>>>(
                fs, featUp, offIn + (size_t)256 * N, 256, H / 2, W / 2);
        }
        // FSM gate: GAP -> 1x1 conv -> sigmoid; armIn = bf16(x*(1+atten))
        gap_kernel<<<Cin, 256, 0, stream>>>(lv[i].feat_l, gapv, N);
        gemv_sigmoid_kernel<<<Cin, 256, 0, stream>>>(lv[i].att_w, gapv, attv, Cin);
        {
            const int tot = Cin * N;
            scale_arm_kernel<<<(tot + 255) / 256, 256, 0, stream>>>(
                lv[i].feat_l, attv, armIn, N, tot);
        }
        // feat_arm = fsm_w @ armIn  (f32 kept for residual; bf16 -> offIn[0:256])
        gemm(fsmB[i], armIn, nullptr, nullptr, featArm, offIn, 256, Cin, H, W, 1, 0);
        // off_feat = off_w @ concat(feat_arm, 2*feat_up)
        gemm(offB[i], offIn, nullptr, nullptr, nullptr, offFeat, 256, 512, H, W, 1, 0);
        // om = conv3x3(off_feat) + om_b   (216 channels: 144 offsets + 72 mask)
        gemm(omWB[i], offFeat, lv[i].om_b, nullptr, omOut, nullptr, 216, 2304, H, W, 3, 0);
        // modulated deformable sampling -> sampB[2304][N]
        {
            dim3 g(N / 256, 72);
            mdcn_sample_kernel<<<g, 256, 0, stream>>>(featUp, omOut, sampB, H, W);
        }
        // f_align = relu(dcn_w @ sampB + dcn_b) + feat_arm   (f32 + bf16 copies)
        gemm(dcnWB[i], sampB, lv[i].dcn_b, featArm, fa32[i], fabf,
             256, 2304, H, W, 1, GF_RELU | GF_ADD);
        // p = conv3x3(f_align) + out_b  -> d_out region
        gemm(outWB[i], fabf, lv[i].out_b, nullptr, pouts[i], nullptr,
             256, 2304, H, W, 3, 0);
    }
}
